// EGNNLayer_11261404250493
// MI455X (gfx1250) — matrix operand order repaired and checked by reading, not compiled, not run on hardware
//
#include <hip/hip_runtime.h>
#include <hip/hip_bf16.h>
#include <stdint.h>

// ---------------------------------------------------------------------------
// EGNN layer fused for MI455X (gfx1250).
// All 5 GEMMs via v_wmma_f32_16x16x32_bf16, 64-row tiles, B-fragment register
// reuse across 4 M-tiles per wave.  Gather path uses GLOBAL_LOAD_ASYNC_TO_LDS
// (ASYNCcnt) when the toolchain exposes the builtin.
// ---------------------------------------------------------------------------

typedef __attribute__((ext_vector_type(16))) __bf16 v16bf;
typedef __attribute__((ext_vector_type(8)))  __bf16 v8bf;
typedef __attribute__((ext_vector_type(8)))  float  v8f;
typedef __attribute__((ext_vector_type(4)))  int    v4i;

#define D_DIM 128
#define LDA   136     // padded LDS row stride (bf16 elems): breaks bank conflicts
#define MROWS 64      // rows (edges / nodes) per block

#if defined(__has_builtin)
#if __has_builtin(__builtin_amdgcn_global_load_async_to_lds_b128)
#define HAVE_ASYNC_LDS 1
#endif
#endif

__device__ __forceinline__ unsigned short f2bf(float f) {
    unsigned int u = __float_as_uint(f);
    unsigned int r = (u + 0x7FFFu + ((u >> 16) & 1u)) >> 16;   // RNE
    return (unsigned short)r;
}
__device__ __forceinline__ float bf2f(unsigned short b) {
    return __uint_as_float(((unsigned int)b) << 16);
}
__device__ __forceinline__ float silu_f(float x) {
    return x / (1.0f + __expf(-x));
}

// 16B global -> LDS copy; async DMA path when available.
__device__ __forceinline__ void copy16_g2l(const void* gsrc, void* ldst) {
#if HAVE_ASYNC_LDS
    __builtin_amdgcn_global_load_async_to_lds_b128(
        (__attribute__((address_space(1))) v4i*)gsrc,
        (__attribute__((address_space(3))) v4i*)ldst, 0, 0);
#else
    *reinterpret_cast<uint4*>(ldst) = *reinterpret_cast<const uint4*>(gsrc);
#endif
}

__device__ __forceinline__ void wait_async_copies() {
#if HAVE_ASYNC_LDS
#if __has_builtin(__builtin_amdgcn_s_wait_asynccnt)
    __builtin_amdgcn_s_wait_asynccnt(0);
#else
    asm volatile("s_wait_asynccnt 0x0" ::: "memory");
#endif
#endif
}

// C[mt] += A(mt-th 16xK tile) * B(Kx16), K = 32*KC, bf16 in, f32 acc.
// B fragment loaded once per K-chunk, reused across MT m-tiles.
template <int KC, int MT>
__device__ __forceinline__ void mmX(const unsigned short* Alds, int ldA,
                                    const unsigned short* BTn, int ldB, int kOff,
                                    int lane, v8f* acc)
{
    const int hf = lane >> 4;          // half-wave select
    const int q  = lane & 15;          // A: row-in-tile | B: col n
    const __bf16* Bbase = reinterpret_cast<const __bf16*>(BTn) + (size_t)q * ldB + kOff;
#pragma unroll
    for (int kb = 0; kb < KC; ++kb) {
        // B fragment: lane = n, halves take K 0-15 / 16-31 -> one 32B vector.
        v16bf b = *reinterpret_cast<const v16bf*>(Bbase + kb * 32 + hf * 16);
#pragma unroll
        for (int mt = 0; mt < MT; ++mt) {
            // A fragment (matches B): lanes 0-15 K 0..15, lanes 16-31 K 16..31
            // -> two contiguous 16B chunks.
            const __bf16* Arow =
                reinterpret_cast<const __bf16*>(Alds) + (mt * 16 + q) * ldA;
            const v8bf* pa = reinterpret_cast<const v8bf*>(Arow + kb * 32 + hf * 16);
            v8bf lo = pa[0];
            v8bf hi = pa[1];                               // +8 elements
            v16bf a = __builtin_shufflevector(lo, hi, 0, 1, 2, 3, 4, 5, 6, 7,
                                                      8, 9, 10, 11, 12, 13, 14, 15);
            acc[mt] = __builtin_amdgcn_wmma_f32_16x16x32_bf16(false, a, false, b,
                                                              (short)0, acc[mt],
                                                              false, false);
        }
    }
}

// ---------------------------------------------------------------------------
// prep kernels
// ---------------------------------------------------------------------------
__global__ void k_init_state(const float* __restrict__ pos,
                             float* __restrict__ agg,
                             float* __restrict__ posOut, int N)
{
    int idx = blockIdx.x * blockDim.x + threadIdx.x;
    if (idx < N * D_DIM) agg[idx] = 0.0f;
    if (idx < N * 3) posOut[idx] = pos[idx];
}

__global__ void k_cvt_h(const float* __restrict__ h,
                        unsigned short* __restrict__ h_bf, int tot)
{
    int idx = blockIdx.x * blockDim.x + threadIdx.x;
    if (idx < tot) h_bf[idx] = f2bf(h[idx]);
}

// dst[n*K + k] = bf16(src[k*Nout + n])   (weights stored transposed)
__global__ void k_transpose_bf16(const float* __restrict__ src,
                                 unsigned short* __restrict__ dst,
                                 int K, int Nout)
{
    int idx = blockIdx.x * blockDim.x + threadIdx.x;
    if (idx < K * Nout) {
        int n = idx / K, k = idx - n * K;
        dst[idx] = f2bf(src[(size_t)k * Nout + n]);
    }
}

// ---------------------------------------------------------------------------
// fused edge kernel: 64 edges/block, 8 waves; wave w owns out cols [16w,16w+16)
// and iterates 4 row-tiles with a register-resident B fragment.
// ---------------------------------------------------------------------------
__global__ __launch_bounds__(256) void egnn_edge_kernel(
    const unsigned short* __restrict__ h_bf,     // [N][128] bf16
    const float*          __restrict__ pos,      // [N][3]
    const long long*      __restrict__ eidx,     // [2][E]
    const unsigned short* __restrict__ W1T,      // [128][256] (We1[0:256]^T)
    const float*          __restrict__ We1last,  // We1[256,:]
    const float*          __restrict__ be1,
    const unsigned short* __restrict__ W2T,      // [128][128]
    const float*          __restrict__ be2,
    const unsigned short* __restrict__ Wc1T,     // [128][128]
    const float*          __restrict__ bc1,
    const float*          __restrict__ Wc2,      // [128]
    float*                __restrict__ agg,      // [N][128] (atomic)
    float*                __restrict__ posOut,   // [N][3]   (atomic)
    int E)
{
    // LDS with lifetime overlays:
    //   Reg1: Ahr (gather..stage1)  ->  Ms (stage2..stage3)
    //   Reg2: Ahc (gather..stage1)  ->  Ts bf16 (stage3..stage4)
    __shared__ alignas(16) unsigned short Reg1[MROWS * LDA];
    __shared__ alignas(16) unsigned short Reg2[MROWS * LDA];
    __shared__ alignas(16) unsigned short Hs  [MROWS * LDA];
    __shared__ int   rows_s[MROWS], cols_s[MROWS];
    __shared__ float dist_s[MROWS], rdist_s[MROWS], diff_s[MROWS][3];

    unsigned short* Ahr = Reg1;
    unsigned short* Ahc = Reg2;
    unsigned short* Ms  = Reg1;
    unsigned short* Tsb = Reg2;

    const int t    = threadIdx.x;
    const int lane = t & 31;
    const int wave = t >> 5;
    const int e0   = blockIdx.x * MROWS;
    if (e0 >= E) return;

    if (t < MROWS) {
        rows_s[t] = (int)eidx[e0 + t];
        cols_s[t] = (int)eidx[(size_t)E + e0 + t];
    }
    __syncthreads();

    // gather 64 h[row] / h[col] rows (bf16, 16B moves; async DMA when present)
#pragma unroll
    for (int i = t; i < MROWS * 16; i += 256) {
        const int m = i >> 4, c = i & 15;
        copy16_g2l(h_bf + (size_t)rows_s[m] * D_DIM + c * 8, Ahr + m * LDA + c * 8);
        copy16_g2l(h_bf + (size_t)cols_s[m] * D_DIM + c * 8, Ahc + m * LDA + c * 8);
    }
    // geometry overlaps the async fill
    if (t < MROWS) {
        int r = rows_s[t], c = cols_s[t];
        float dx = pos[r * 3 + 0] - pos[c * 3 + 0];
        float dy = pos[r * 3 + 1] - pos[c * 3 + 1];
        float dz = pos[r * 3 + 2] - pos[c * 3 + 2];
        float dd = fmaxf(sqrtf(dx * dx + dy * dy + dz * dz), 1e-6f);
        dist_s[t] = dd; rdist_s[t] = 1.0f / dd;
        diff_s[t][0] = dx; diff_s[t][1] = dy; diff_s[t][2] = dz;
    }

    const int n0 = wave * 16;
    const int hf = lane >> 4, q = lane & 15;
    const int n  = n0 + q;

    // warm L2/L0 for later-stage weight slices (global_prefetch_b8)
    __builtin_prefetch(W2T  + (size_t)n * 128 + hf * 64, 0, 1);
    __builtin_prefetch(Wc1T + (size_t)n * 128 + hf * 64, 0, 1);

    wait_async_copies();
    __syncthreads();

    // stage 1: hidden1 = silu(h_r@We1a + h_c@We1b + dist*We1[256] + be1)
    {
        v8f acc[4] = {};
        mmX<4, 4>(Ahr, LDA, W1T + (size_t)n0 * 256, 256, 0,   lane, acc);
        mmX<4, 4>(Ahc, LDA, W1T + (size_t)n0 * 256, 256, 128, lane, acc);
        const float wl = We1last[n], b = be1[n];
#pragma unroll
        for (int mt = 0; mt < 4; ++mt)
#pragma unroll
            for (int r = 0; r < 8; ++r) {
                int m = mt * 16 + r + hf * 8;
                float x = acc[mt][r] + dist_s[m] * wl + b;
                Hs[m * LDA + n] = f2bf(silu_f(x));
            }
    }
    __syncthreads();

    // stage 2: msg = silu(hidden1 @ We2 + be2); scatter-add into agg
    {
        v8f acc[4] = {};
        mmX<4, 4>(Hs, LDA, W2T + (size_t)n0 * 128, 128, 0, lane, acc);
        const float b = be2[n];
#pragma unroll
        for (int mt = 0; mt < 4; ++mt)
#pragma unroll
            for (int r = 0; r < 8; ++r) {
                int m = mt * 16 + r + hf * 8;
                float v = silu_f(acc[mt][r] + b);
                atomicAdd(agg + (size_t)rows_s[m] * D_DIM + n, v);
                Ms[m * LDA + n] = f2bf(v);
            }
    }
    __syncthreads();

    // stage 3: tcoord = silu(msg @ Wc1 + bc1)  (bf16 into Reg2)
    {
        v8f acc[4] = {};
        mmX<4, 4>(Ms, LDA, Wc1T + (size_t)n0 * 128, 128, 0, lane, acc);
        const float b = bc1[n];
#pragma unroll
        for (int mt = 0; mt < 4; ++mt)
#pragma unroll
            for (int r = 0; r < 8; ++r) {
                int m = mt * 16 + r + hf * 8;
                Tsb[m * LDA + n] = f2bf(silu_f(acc[mt][r] + b));
            }
    }
    __syncthreads();

    // stage 4: coord_weight = clip(tcoord @ Wc2, -1, 1); pos scatter-add
    if (t < MROWS) {
        float w = 0.0f;
#pragma unroll 8
        for (int k = 0; k < D_DIM; ++k) w += bf2f(Tsb[t * LDA + k]) * Wc2[k];
        w = fminf(fmaxf(w, -1.0f), 1.0f);
        float s = w * rdist_s[t];
        int r = rows_s[t];
        atomicAdd(posOut + (size_t)r * 3 + 0, diff_s[t][0] * s);
        atomicAdd(posOut + (size_t)r * 3 + 1, diff_s[t][1] * s);
        atomicAdd(posOut + (size_t)r * 3 + 2, diff_s[t][2] * s);
    }
}

// ---------------------------------------------------------------------------
// node kernel: h_out = h + silu(concat(h, agg) @ Wn1 + bn1) @ Wn2 + bn2
// 64 nodes/block with clamped-gather tail handling (EXEC uniform thru WMMA).
// ---------------------------------------------------------------------------
__global__ __launch_bounds__(256) void egnn_node_kernel(
    const float*          __restrict__ h,        // [N][128] f32
    const unsigned short* __restrict__ h_bf,     // [N][128] bf16
    const float*          __restrict__ agg,      // [N][128] f32
    const unsigned short* __restrict__ Wn1T,     // [128][256]
    const float*          __restrict__ bn1,
    const unsigned short* __restrict__ Wn2T,     // [128][128]
    const float*          __restrict__ bn2,
    float*                __restrict__ outH,     // [N][128]
    int N)
{
    __shared__ alignas(16) unsigned short Ah[MROWS * LDA];
    __shared__ alignas(16) unsigned short Ag[MROWS * LDA];
    __shared__ alignas(16) unsigned short Hn[MROWS * LDA];

    const int t    = threadIdx.x;
    const int lane = t & 31;
    const int wave = t >> 5;
    const int b0   = blockIdx.x * MROWS;
    if (b0 >= N) return;

    // bf16 copy of 64 h rows (async DMA) + f32->bf16 convert of 64 agg rows
#pragma unroll
    for (int i = t; i < MROWS * 16; i += 256) {
        const int m = i >> 4, c = i & 15;
        const int node = min(b0 + m, N - 1);
        copy16_g2l(h_bf + (size_t)node * D_DIM + c * 8, Ah + m * LDA + c * 8);
        const float* ga = agg + (size_t)node * D_DIM + c * 8;
#pragma unroll
        for (int i8 = 0; i8 < 8; ++i8)
            Ag[m * LDA + c * 8 + i8] = f2bf(ga[i8]);
    }
    wait_async_copies();
    __syncthreads();

    const int n0 = wave * 16;
    const int hf = lane >> 4, q = lane & 15;
    const int n  = n0 + q;

    __builtin_prefetch(Wn2T + (size_t)n * 128 + hf * 64, 0, 1);

    // hidden = silu(concat(h, agg) @ Wn1 + bn1)   (K = 256)
    {
        v8f acc[4] = {};
        mmX<4, 4>(Ah, LDA, Wn1T + (size_t)n0 * 256, 256, 0,   lane, acc);
        mmX<4, 4>(Ag, LDA, Wn1T + (size_t)n0 * 256, 256, 128, lane, acc);
        const float b = bn1[n];
#pragma unroll
        for (int mt = 0; mt < 4; ++mt)
#pragma unroll
            for (int r = 0; r < 8; ++r) {
                int m = mt * 16 + r + hf * 8;
                Hn[m * LDA + n] = f2bf(silu_f(acc[mt][r] + b));
            }
    }
    __syncthreads();

    // out = h + hidden @ Wn2 + bn2
    {
        v8f acc[4] = {};
        mmX<4, 4>(Hn, LDA, Wn2T + (size_t)n0 * 128, 128, 0, lane, acc);
        const float b = bn2[n];
#pragma unroll
        for (int mt = 0; mt < 4; ++mt)
#pragma unroll
            for (int r = 0; r < 8; ++r) {
                int m = mt * 16 + r + hf * 8;
                if (b0 + m < N) {
                    size_t o = (size_t)(b0 + m) * D_DIM + n;
                    outH[o] = h[o] + acc[mt][r] + b;
                }
            }
    }
}

// ---------------------------------------------------------------------------
// launch
// ---------------------------------------------------------------------------
extern "C" void kernel_launch(void* const* d_in, const int* in_sizes, int n_in,
                              void* d_out, int out_size, void* d_ws, size_t ws_size,
                              hipStream_t stream)
{
    (void)n_in; (void)out_size; (void)ws_size;
    const float*     h    = (const float*)d_in[0];
    const float*     pos  = (const float*)d_in[1];
    const long long* eidx = (const long long*)d_in[2];
    const float*     We1  = (const float*)d_in[3];
    const float*     be1  = (const float*)d_in[4];
    const float*     We2  = (const float*)d_in[5];
    const float*     be2  = (const float*)d_in[6];
    const float*     Wn1  = (const float*)d_in[7];
    const float*     bn1  = (const float*)d_in[8];
    const float*     Wn2  = (const float*)d_in[9];
    const float*     bn2  = (const float*)d_in[10];
    const float*     Wc1  = (const float*)d_in[11];
    const float*     bc1  = (const float*)d_in[12];
    const float*     Wc2  = (const float*)d_in[13];

    const int N = in_sizes[0] / D_DIM;     // 50000
    const int E = in_sizes[2] / 2;         // 800000

    float* outH   = (float*)d_out;                       // [N][128]
    float* posOut = (float*)d_out + (size_t)N * D_DIM;   // [N][3]

    // workspace layout
    uint8_t* ws = (uint8_t*)d_ws;
    unsigned short* h_bf = (unsigned short*)ws;                       // N*128*2
    float*          agg  = (float*)(ws + (size_t)N * D_DIM * 2);     // N*128*4
    uint8_t* wsw = ws + (size_t)N * D_DIM * 2 + (size_t)N * D_DIM * 4;
    unsigned short* W1T  = (unsigned short*)(wsw);                   // 128*256
    unsigned short* W2T  = (unsigned short*)(wsw + 65536);           // 128*128
    unsigned short* Wc1T = (unsigned short*)(wsw + 98304);           // 128*128
    unsigned short* Wn1T = (unsigned short*)(wsw + 131072);          // 128*256
    unsigned short* Wn2T = (unsigned short*)(wsw + 196608);          // 128*128

    const int TB = 256;
    int tot = N * D_DIM;

    k_init_state<<<(tot + TB - 1) / TB, TB, 0, stream>>>(pos, agg, posOut, N);
    k_cvt_h<<<(tot + TB - 1) / TB, TB, 0, stream>>>(h, h_bf, tot);
    k_transpose_bf16<<<(256 * 128 + TB - 1) / TB, TB, 0, stream>>>(We1, W1T, 256, 128);
    k_transpose_bf16<<<(128 * 128 + TB - 1) / TB, TB, 0, stream>>>(We2, W2T, 128, 128);
    k_transpose_bf16<<<(128 * 128 + TB - 1) / TB, TB, 0, stream>>>(Wc1, Wc1T, 128, 128);
    k_transpose_bf16<<<(256 * 128 + TB - 1) / TB, TB, 0, stream>>>(Wn1, Wn1T, 256, 128);
    k_transpose_bf16<<<(128 * 128 + TB - 1) / TB, TB, 0, stream>>>(Wn2, Wn2T, 128, 128);

    egnn_edge_kernel<<<(E + MROWS - 1) / MROWS, TB, 0, stream>>>(
        h_bf, pos, eidx,
        W1T, We1 + 256 * 128, be1,
        W2T, be2, Wc1T, bc1, Wc2,
        agg, posOut, E);

    egnn_node_kernel<<<(N + MROWS - 1) / MROWS, TB, 0, stream>>>(
        h, h_bf, agg, Wn1T, bn1, Wn2T, bn2, outH, N);
}
